// sLSTMLayer_49787260895632
// MI455X (gfx1250) — matrix operand order repaired and checked by reading, not compiled, not run on hardware
//
#include <hip/hip_runtime.h>
#include <hip/hip_bf16.h>
#include <stdint.h>

// ---- problem constants (match setup_inputs) ----
#define B_   4
#define S_   2048
#define D_   1024
#define NH_  4
#define DH_  256
#define KC_  4
#define M_   (B_ * S_)          // 8192 rows for the GEMMs
#define C_   32                 // scan chunks per sequence
#define L_   (S_ / C_)          // 64 steps per chunk

typedef __bf16 bf16_t;
typedef __attribute__((ext_vector_type(16))) bf16_t bf16x16;
typedef __attribute__((ext_vector_type(8)))  bf16_t bf16x8;
typedef __attribute__((ext_vector_type(8)))  float  floatx8;
typedef __attribute__((ext_vector_type(4)))  float  floatx4;

__device__ __forceinline__ bf16_t f2bf(float f) {
    unsigned u = __builtin_bit_cast(unsigned, f);
    unsigned r = u + 0x7FFFu + ((u >> 16) & 1u);   // round-to-nearest-even
    unsigned short h = (unsigned short)(r >> 16);
    return __builtin_bit_cast(bf16_t, h);
}

__device__ __forceinline__ float sigm(float x) { return 1.0f / (1.0f + __expf(-x)); }

// ---------------------------------------------------------------------------
// Kernel 0: convert the four (NH,DH,DH) fp32 weights to one bf16 buffer
// Wb layout: [gate][h][o][i], gate order z,i,f,o
// ---------------------------------------------------------------------------
__global__ __launch_bounds__(256) void wconv_kernel(
    const float* __restrict__ Wz, const float* __restrict__ Wi,
    const float* __restrict__ Wf, const float* __restrict__ Wo,
    bf16_t* __restrict__ Wb)
{
    int idx = blockIdx.x * 256 + threadIdx.x;          // 0 .. 4*NH*DH*DH-1
    int g   = idx >> 18;                               // NH*DH*DH = 262144
    int r   = idx & 262143;
    const float* src = (g == 0) ? Wz : (g == 1) ? Wi : (g == 2) ? Wf : Wo;
    Wb[idx] = f2bf(src[r]);
}

// ---------------------------------------------------------------------------
// Kernel 1: causal depthwise conv (K=4) + SiLU, emit bf16 activations
// ---------------------------------------------------------------------------
__global__ __launch_bounds__(256) void conv_silu_kernel(
    const float* __restrict__ x, const float* __restrict__ ck,
    bf16_t* __restrict__ xconv)
{
    int idx = blockIdx.x * 256 + threadIdx.x;          // 0 .. B*S*D-1
    int d   = idx & (D_ - 1);
    int s   = (idx / D_) & (S_ - 1);
    const float* kr = ck + d * KC_;                    // conv_kernel (D,1,K)
    float acc = 0.0f;
#pragma unroll
    for (int k = 0; k < KC_; ++k) {
        int sp = s - (KC_ - 1) + k;
        if (sp >= 0) acc = fmaf(x[idx + (sp - s) * D_], kr[k], acc);
    }
    float y = acc * sigm(acc);                          // SiLU
    xconv[idx] = f2bf(y);
}

// ---------------------------------------------------------------------------
// Kernel 2: 4-gate head-wise GEMM on the WMMA engine.
//   grid = (M/64, N-half, head), block = 256 threads = 8 waves.
//   64x256 bf16 A strip staged in LDS (32 KB), shared by all 8 waves.
//   Each wave: 4 M-tiles x 1 N-tile x 4 gates -> each B fragment feeds
//   4 WMMAs (4x less weight traffic from L2 than 1 M-tile blocking).
//   K=256 in 8 steps of 32. Epilogue: tanh(z)/sigmoid(o), float4 store.
// ---------------------------------------------------------------------------
__global__ __launch_bounds__(256) void gemm_gates_kernel(
    const bf16_t* __restrict__ xconv, const bf16_t* __restrict__ Wb,
    floatx4* __restrict__ gates)
{
    __shared__ bf16_t sA[64 * DH_];                    // 32 KB

    const int t     = threadIdx.x;
    const int lane  = t & 31;
    const int wave  = t >> 5;
    const int lh    = lane >> 4;                       // lane-half (0/1)
    const int row   = lane & 15;
    const int m0    = blockIdx.x * 64;                 // 0..127 strips
    const int nh0   = blockIdx.y * 128;                // N half (0 or 128)
    const int h     = blockIdx.z;                      // head

    // stage A strip: rows m0..m0+63, cols h*DH .. h*DH+255 of xconv
#pragma unroll
    for (int j = 0; j < 8; ++j) {
        int c  = j * 256 + t;                          // 2048 chunks of 8 bf16
        int r  = c >> 5;
        int i0 = (c & 31) << 3;
        *(bf16x8*)(sA + r * DH_ + i0) =
            *(const bf16x8*)(xconv + (size_t)(m0 + r) * D_ + h * DH_ + i0);
    }
    __syncthreads();

    floatx8 acc[4][4];                                 // [gate][mtile]
#pragma unroll
    for (int g = 0; g < 4; ++g)
#pragma unroll
        for (int mt = 0; mt < 4; ++mt)
#pragma unroll
            for (int e = 0; e < 8; ++e) acc[g][mt][e] = 0.0f;

    const int ncol = nh0 + wave * 16 + row;            // this wave's N column

#pragma unroll
    for (int ks = 0; ks < 8; ++ks) {
        // A fragments (16x32 bf16) for the 4 M-tiles, from LDS
        const int kbA = ks * 32 + lh * 8;
        bf16x16 a[4];
#pragma unroll
        for (int mt = 0; mt < 4; ++mt) {
            bf16x8 alo = *(const bf16x8*)(sA + (mt * 16 + row) * DH_ + kbA);
            bf16x8 ahi = *(const bf16x8*)(sA + (mt * 16 + row) * DH_ + kbA + 16);
#pragma unroll
            for (int e = 0; e < 8; ++e) { a[mt][e] = alo[e]; a[mt][e + 8] = ahi[e]; }
        }

        const int kbB = ks * 32 + lh * 8;              // B: K = lh*8 + e, +16 for e>=8
#pragma unroll
        for (int g = 0; g < 4; ++g) {
            const bf16_t* wrow =
                Wb + (((size_t)g * NH_ + h) * DH_ + ncol) * DH_ + kbB;
            bf16x8 blo = *(const bf16x8*)(wrow);
            bf16x8 bhi = *(const bf16x8*)(wrow + 16);
            bf16x16 bf;
#pragma unroll
            for (int e = 0; e < 8; ++e) { bf[e] = blo[e]; bf[e + 8] = bhi[e]; }
#pragma unroll
            for (int mt = 0; mt < 4; ++mt) {
                acc[g][mt] = __builtin_amdgcn_wmma_f32_16x16x32_bf16(
                    false, a[mt], false, bf, (short)0, acc[g][mt], false, false);
            }
        }
    }

    // epilogue: activations + interleaved float4 store
#pragma unroll
    for (int mt = 0; mt < 4; ++mt) {
#pragma unroll
        for (int r = 0; r < 8; ++r) {
            const int mrow = m0 + mt * 16 + lh * 8 + r;  // C/D layout: M = r + 8*lh
            floatx4 v;
            v.x = tanhf(acc[0][mt][r]);                // z
            v.y = acc[1][mt][r];                       // i (log-space)
            v.z = acc[2][mt][r];                       // f (log-space)
            v.w = sigm(acc[3][mt][r]);                 // o
            gates[(size_t)mrow * D_ + h * DH_ + ncol] = v;
        }
    }
}

// ---------------------------------------------------------------------------
// Scan, pass 1: per-chunk local summaries.
//   One block per (b,h,chunk) = 512 blocks x 256 threads (one per channel d).
//   Local scan with m-hat init = -inf (handled analytically at t=0):
//     F = sum(f), M = m-hat_L, Chat, Nhat.
// ---------------------------------------------------------------------------
__global__ __launch_bounds__(256) void scan_pass1_kernel(
    const floatx4* __restrict__ gates, floatx4* __restrict__ summ)
{
    const int blk = blockIdx.x;                        // (b*NH + h)*C + j
    const int j   = blk & (C_ - 1);
    const int bh  = blk / C_;
    const int b   = bh >> 2;
    const int h   = bh & 3;
    const int d   = threadIdx.x;
    const int hd  = h * DH_ + d;

    const floatx4* p = gates + (size_t)(b * S_ + j * L_) * D_ + hd;

    floatx4 g = p[0];                                  // t = 0 closed form
    float F  = g.z;                                    // sum of f
    float mh = g.y;                                    // m-hat_1 = i_1
    float ch = g.x;                                    // c-hat_1 = z_1
    float nh = 1.0f;                                   // n-hat_1 = 1
    for (int t = 1; t < L_; ++t) {
        if (t + 8 < L_) __builtin_prefetch(p + (size_t)(t + 8) * D_, 0, 0);
        floatx4 gn = p[(size_t)t * D_];
        const float mn = fmaxf(gn.z + mh, gn.y);
        const float ih = __expf(gn.y - mn);
        const float fh = __expf(gn.z + mh - mn);
        ch = fh * ch + ih * gn.x;
        nh = fh * nh + ih;
        mh = mn;
        F += gn.z;
    }
    floatx4 s; s.x = ch; s.y = nh; s.z = mh; s.w = F;
    summ[(size_t)blk * DH_ + d] = s;
}

// ---------------------------------------------------------------------------
// Scan, middle: serially compose the C=32 chunk summaries per channel,
// recording the exact incoming state (c,n,m) of every chunk.
//   4096 threads total (one per (b,h,d)); 32 tiny steps each.
// ---------------------------------------------------------------------------
__global__ __launch_bounds__(256) void scan_mid_kernel(
    const floatx4* __restrict__ summ, floatx4* __restrict__ states)
{
    const int tid = blockIdx.x * 256 + threadIdx.x;    // 0..4095
    const int d   = tid & (DH_ - 1);
    const int bh  = tid >> 8;                          // b*NH + h

    float c = 0.0f, n = 0.0f, m = 0.0f;                // reference init
#pragma unroll 4
    for (int j = 0; j < C_; ++j) {
        const size_t idx = ((size_t)bh * C_ + j) * DH_ + d;
        floatx4 st; st.x = c; st.y = n; st.z = m; st.w = 0.0f;
        states[idx] = st;                              // state entering chunk j
        const floatx4 s = summ[idx];                   // {Chat, Nhat, M, F}
        const float mp = fmaxf(m + s.w, s.z);
        const float ea = __expf(m + s.w - mp);
        const float eb = __expf(s.z - mp);
        c = ea * c + eb * s.x;
        n = ea * n + eb * s.y;
        m = mp;
    }
}

// ---------------------------------------------------------------------------
// Scan, pass 2: exact re-scan of each chunk from its incoming state, fused
// with the multi-head group norm (block = all 256 channels of one head).
//   512 blocks; serial depth is only L_=64 steps.
// ---------------------------------------------------------------------------
__global__ __launch_bounds__(256) void scan_pass2_kernel(
    const floatx4* __restrict__ gates, const floatx4* __restrict__ states,
    const float* __restrict__ gamma, float* __restrict__ out)
{
    __shared__ float red[16];
    const int blk  = blockIdx.x;                       // (b*NH + h)*C + j
    const int j    = blk & (C_ - 1);
    const int bh   = blk / C_;
    const int b    = bh >> 2;
    const int h    = bh & 3;
    const int d    = threadIdx.x;
    const int lane = d & 31;
    const int wave = d >> 5;
    const int hd   = h * DH_ + d;
    const float scale = 1.0f + gamma[hd];

    const floatx4 st = states[(size_t)blk * DH_ + d];
    float c = st.x, n = st.y, m = st.z;

    const floatx4* p = gates + (size_t)(b * S_ + j * L_) * D_ + hd;
    float* po        = out   + (size_t)(b * S_ + j * L_) * D_ + hd;

    floatx4 g = p[0];
    for (int t = 0; t < L_; ++t) {
        if (t + 4 < L_) __builtin_prefetch(p + (size_t)(t + 4) * D_, 0, 0);
        floatx4 gn = (t + 1 < L_) ? p[(size_t)(t + 1) * D_] : g;

        const float zt = g.x, it = g.y, ft = g.z, ot = g.w;
        const float mn = fmaxf(ft + m, it);
        const float ih = __expf(it - mn);
        const float fh = __expf(ft + m - mn);
        c = fh * c + ih * zt;
        n = fh * n + ih;
        m = mn;
        const float hv = ot * (c / (n + 1e-8f));

        // block-wide mean/var over DH
        float s1 = hv, s2 = hv * hv;
#pragma unroll
        for (int off = 16; off; off >>= 1) {
            s1 += __shfl_xor(s1, off, 32);
            s2 += __shfl_xor(s2, off, 32);
        }
        if (lane == 0) { red[wave * 2] = s1; red[wave * 2 + 1] = s2; }
        __syncthreads();
        float t1 = 0.0f, t2 = 0.0f;
#pragma unroll
        for (int wv = 0; wv < 8; ++wv) { t1 += red[wv * 2]; t2 += red[wv * 2 + 1]; }
        const float mean = t1 * (1.0f / DH_);
        const float var  = t2 * (1.0f / DH_) - mean * mean;
        const float rstd = rsqrtf(var + 1e-5f);
        po[(size_t)t * D_] = (hv - mean) * rstd * scale;
        __syncthreads();                               // LDS reuse next step
        g = gn;
    }
}

// ---------------------------------------------------------------------------
// host-side orchestration
// ---------------------------------------------------------------------------
extern "C" void kernel_launch(void* const* d_in, const int* in_sizes, int n_in,
                              void* d_out, int out_size, void* d_ws, size_t ws_size,
                              hipStream_t stream)
{
    (void)in_sizes; (void)n_in; (void)out_size; (void)ws_size;

    const float* x     = (const float*)d_in[0];
    const float* ck    = (const float*)d_in[1];
    const float* Wz    = (const float*)d_in[2];
    const float* Wi    = (const float*)d_in[3];
    const float* Wf    = (const float*)d_in[4];
    const float* Wo    = (const float*)d_in[5];
    const float* gamma = (const float*)d_in[6];
    float* out = (float*)d_out;

    // workspace layout (all 256B aligned):
    //   [0,    2MB) : Wb     (4*NH*DH*DH bf16)
    //   [2MB, 18MB) : xconv  (B*S*D bf16)
    //   [18MB,146MB): gates  (B*S*D float4 {z,i,f,o})
    //   [146MB,148MB): summ  (B*NH*C*DH float4 {Chat,Nhat,M,F})
    //   [148MB,150MB): states(B*NH*C*DH float4 {c,n,m,-})
    char* ws = (char*)d_ws;
    bf16_t*  Wb     = (bf16_t*)ws;
    bf16_t*  xconv  = (bf16_t*)(ws + (size_t)2   * 1024 * 1024);
    floatx4* gts    = (floatx4*)(ws + (size_t)18  * 1024 * 1024);
    floatx4* summ   = (floatx4*)(ws + (size_t)146 * 1024 * 1024);
    floatx4* states = (floatx4*)(ws + (size_t)148 * 1024 * 1024);

    wconv_kernel<<<(4 * NH_ * DH_ * DH_) / 256, 256, 0, stream>>>(Wz, Wi, Wf, Wo, Wb);
    conv_silu_kernel<<<(B_ * S_ * D_) / 256, 256, 0, stream>>>(x, ck, xconv);
    gemm_gates_kernel<<<dim3(M_ / 64, 2, NH_), 256, 0, stream>>>(xconv, Wb, gts);
    scan_pass1_kernel<<<B_ * NH_ * C_, 256, 0, stream>>>(gts, summ);
    scan_mid_kernel<<<(B_ * NH_ * DH_) / 256, 256, 0, stream>>>(summ, states);
    scan_pass2_kernel<<<B_ * NH_ * C_, 256, 0, stream>>>(gts, states, gamma, out);
}
